// TTRL_74904229643130
// MI455X (gfx1250) — matrix operand order repaired and checked by reading, not compiled, not run on hardware
//
#include <hip/hip_runtime.h>
#include <hip/hip_bf16.h>
#include <stddef.h>

// ---------------------------------------------------------------------------
// MI455X (gfx1250) TTRL kernel.
// Strategy: pre-collapse TT chains once (tiny), then memory-bound streaming
// GEMMs with v_wmma_f32_16x16x32_bf16 (f32 accumulate). ~150MB HBM traffic
// dominates (~6.5us @ 23.3TB/s); bf16 WMMA compute is ~1-2us.
// ---------------------------------------------------------------------------

typedef __bf16 bf16_t;
typedef bf16_t v16bf __attribute__((ext_vector_type(16)));
typedef float  v8f   __attribute__((ext_vector_type(8)));

union FragBF {
    v16bf v;
    unsigned short u[16];
    uint4 q[2];
};

// float -> bf16 bits, round-to-nearest-even
__device__ __forceinline__ unsigned short f2bf(float f) {
    unsigned u = __float_as_uint(f);
    u += 0x7FFFu + ((u >> 16) & 1u);
    return (unsigned short)(u >> 16);
}

// A-fragment (16x32, MxK, 16-bit): per ISA table, lane L<16 holds M=L,
// K = {0..7} U {16..23}; lane L>=16 holds M=L-16, K = {8..15} U {24..31}.
// src is row-major [M][K] bf16 bits with leading dim ld (elements).
__device__ __forceinline__ v16bf load_frag_a(const unsigned short* __restrict__ src,
                                             int ld, int m0, int k0, int lane) {
    const int m  = m0 + (lane & 15);
    const int kb = k0 + ((lane >> 4) << 3);
    const unsigned short* p = src + (size_t)m * ld + kb;
    FragBF f;
    f.q[0] = *(const uint4*)(p);        // K = kb .. kb+7
    f.q[1] = *(const uint4*)(p + 16);   // K = kb+16 .. kb+23
    return f.v;
}

// B-fragment (32x16, KxN, 16-bit) loaded from TRANSPOSED storage srcT[N][K]:
// lane L holds N = L&15, K = 8*(L>=16) + {0..7} U {16..23} (matches A).
__device__ __forceinline__ v16bf load_frag_bt(const unsigned short* __restrict__ srcT,
                                              int ld, int k0, int n0, int lane) {
    const int n  = n0 + (lane & 15);
    const int kb = k0 + ((lane >> 4) << 3);
    const unsigned short* p = srcT + (size_t)n * ld + kb;
    FragBF f;
    f.q[0] = *(const uint4*)(p);
    f.q[1] = *(const uint4*)(p + 16);
    return f.v;
}

__device__ __forceinline__ v8f wmma_bf16(v16bf a, v16bf b, v8f c) {
    // (neg_a, A, neg_b, B, c_mod, C, reuse_a, reuse_b)
    return __builtin_amdgcn_wmma_f32_16x16x32_bf16(false, a, false, b,
                                                   (short)0, c, false, false);
}

// ---------------------------------------------------------------------------
// Kernel 1: collapse TT chains into dense bf16 operand matrices (runs once,
// single workgroup; all work is tiny).
//   g1t[r1][n123]  (32 x 512)   = (f0*f1*f2)^T
//   g2t[r2][n45]   (32 x 64)    = (f3*f4)^T
//   w3 [o][s]      (4096 x 32)  = g0*g1*g2
// ---------------------------------------------------------------------------
__global__ void tt_precompute_kernel(const float* __restrict__ f0,
                                     const float* __restrict__ f1,
                                     const float* __restrict__ f2,
                                     const float* __restrict__ f3,
                                     const float* __restrict__ f4,
                                     const float* __restrict__ gg0,
                                     const float* __restrict__ gg1,
                                     const float* __restrict__ gg2,
                                     unsigned short* __restrict__ g1t,
                                     unsigned short* __restrict__ g2t,
                                     unsigned short* __restrict__ w3) {
    __shared__ float h1[64][32];    // (n1n2, p2)
    __shared__ float w2[256][32];   // (o0o1, q2)
    const int t = threadIdx.x;

    // h1[n1n2][p2] = sum_p1 f0[n1][p1] * f1[p1][n2][p2]
    for (int i = t; i < 64 * 32; i += 256) {
        int a = i >> 5, p2 = i & 31;
        int n1 = a >> 3, n2 = a & 7;
        float s = 0.f;
        for (int p1 = 0; p1 < 32; ++p1)
            s = fmaf(f0[n1 * 32 + p1], f1[(p1 * 8 + n2) * 32 + p2], s);
        h1[a][p2] = s;
    }
    // w2[o0o1][q2] = sum_q1 g0[o0][q1] * g1[q1][o1][q2]
    for (int i = t; i < 256 * 32; i += 256) {
        int o01 = i >> 5, q2 = i & 31;
        int o0 = o01 >> 4, o1 = o01 & 15;
        float s = 0.f;
        for (int q1 = 0; q1 < 32; ++q1)
            s = fmaf(gg0[o0 * 32 + q1], gg1[(q1 * 16 + o1) * 32 + q2], s);
        w2[o01][q2] = s;
    }
    __syncthreads();
    // g1t[r][n123] = sum_p2 h1[n1n2][p2] * f2[p2][n3][r]
    for (int i = t; i < 512 * 32; i += 256) {
        int n123 = i >> 5, r = i & 31;
        int n12 = n123 >> 3, n3 = n123 & 7;
        float s = 0.f;
        for (int p2 = 0; p2 < 32; ++p2)
            s = fmaf(h1[n12][p2], f2[(p2 * 8 + n3) * 32 + r], s);
        g1t[r * 512 + n123] = f2bf(s);
    }
    // g2t[r][n45] = sum_p f3[n4][p] * f4[p][n5][r]
    for (int i = t; i < 64 * 32; i += 256) {
        int n45 = i >> 5, r = i & 31;
        int n4 = n45 >> 3, n5 = n45 & 7;
        float s = 0.f;
        for (int p = 0; p < 32; ++p)
            s = fmaf(f3[n4 * 32 + p], f4[(p * 8 + n5) * 32 + r], s);
        g2t[r * 64 + n45] = f2bf(s);
    }
    // w3[o][s] = sum_q2 w2[o0o1][q2] * g2[q2][o2][s]
    for (int i = t; i < 4096 * 32; i += 256) {
        int o = i >> 5, so = i & 31;
        int o01 = o >> 4, o2 = o & 15;
        float s = 0.f;
        for (int q2 = 0; q2 < 32; ++q2)
            s = fmaf(w2[o01][q2], gg2[(q2 * 16 + o2) * 32 + so], s);
        w3[o * 32 + so] = f2bf(s);
    }
}

// ---------------------------------------------------------------------------
// Kernel 2: per-sample contraction.  One WG per sample (128 thr = 4 waves).
//   Y'(32x64) = G1^T(32x512) * X_b(512x64)   -- K streamed in 4 chunks of 128
//   A (32x32) = Y'(32x64)    * G2(64x32)
//   z[s]      = vec(A)(1024) . core(1024x32)  (VALU; tiny)
// ---------------------------------------------------------------------------
__global__ void __launch_bounds__(128)
tt_contract_kernel(const float* __restrict__ x,
                   const float* __restrict__ core,
                   const unsigned short* __restrict__ g1t,
                   const unsigned short* __restrict__ g2t,
                   unsigned short* __restrict__ zbf) {
    __shared__ unsigned short xbT[64][136];  // transposed chunk, padded (16B-aligned rows)
    __shared__ unsigned short ybf[32][64];   // Y' in bf16 (A-operand of phase 2)
    __shared__ float abuf[1024];             // A, row-major (r1*32 + r2)
    __shared__ float zpart[4][32];

    const int b    = blockIdx.x;
    const int t    = threadIdx.x;
    const int lane = t & 31;
    const int w    = t >> 5;
    const float* xb = x + (size_t)b * 32768;

    // Per-thread loop-invariant bases for the stream+transpose:
    //   global: thread reads floats t*4 + it*512  (float4 each, coalesced)
    //   LDS:    column nn=(t&15)*4 is loop-invariant; row kr=(t>>4)+it*8
    //   => all 16 loads / 64 stores use constant immediate offsets.
    const float* xrow = xb + t * 4;
    unsigned short* ldsb = &xbT[(t & 15) * 4][t >> 4];

    // Wave w owns D tiles (mt = w>>1, nt in {2*(w&1), 2*(w&1)+1}) of Y'.
    const int mt  = w >> 1;
    const int ntA = (w & 1) * 2;
    v8f acc0 = {0.f, 0.f, 0.f, 0.f, 0.f, 0.f, 0.f, 0.f};
    v8f acc1 = {0.f, 0.f, 0.f, 0.f, 0.f, 0.f, 0.f, 0.f};

    // ---- Phase 1: Y' = G1^T * X_b, K=512 in 4 chunks of 128 ----
    for (int kc = 0; kc < 4; ++kc) {
        const float* src = xrow + kc * 8192;
        if (kc < 3)
            __builtin_prefetch(src + 8192, 0, 3);  // near-scope prefetch of next chunk
        __syncthreads();
        #pragma unroll
        for (int it = 0; it < 16; ++it) {
            const float4 vv = *(const float4*)(src + it * 512);
            ldsb[it * 8 + 0 * 136] = f2bf(vv.x);
            ldsb[it * 8 + 1 * 136] = f2bf(vv.y);
            ldsb[it * 8 + 2 * 136] = f2bf(vv.z);
            ldsb[it * 8 + 3 * 136] = f2bf(vv.w);
        }
        __syncthreads();
        #pragma unroll
        for (int ks = 0; ks < 4; ++ks) {
            const int kglob = kc * 128 + ks * 32;
            v16bf af = load_frag_a(g1t, 512, mt * 16, kglob, lane);     // L2-hot
            v16bf b0 = load_frag_bt(&xbT[0][0], 136, ks * 32, ntA * 16, lane);
            v16bf b1 = load_frag_bt(&xbT[0][0], 136, ks * 32, ntA * 16 + 16, lane);
            acc0 = wmma_bf16(af, b0, acc0);
            acc1 = wmma_bf16(af, b1, acc1);
        }
    }
    {   // store Y' as bf16 (C/D layout: n = lane&15, m = v + 8*(lane>=16))
        const int nn = lane & 15;
        const int mr = mt * 16 + ((lane >> 4) << 3);
        #pragma unroll
        for (int v = 0; v < 8; ++v) {
            ybf[mr + v][ntA * 16 + nn]       = f2bf(acc0[v]);
            ybf[mr + v][(ntA + 1) * 16 + nn] = f2bf(acc1[v]);
        }
    }
    __syncthreads();

    // ---- Phase 2: A = Y' * G2 (4 tiles, one per wave; K=64) ----
    {
        const int mt2 = w >> 1, nt2 = w & 1;
        v8f acc = {0.f, 0.f, 0.f, 0.f, 0.f, 0.f, 0.f, 0.f};
        #pragma unroll
        for (int ks = 0; ks < 2; ++ks) {
            v16bf af = load_frag_a(&ybf[0][0], 64, mt2 * 16, ks * 32, lane);
            v16bf bf = load_frag_bt(g2t, 64, ks * 32, nt2 * 16, lane);
            acc = wmma_bf16(af, bf, acc);
        }
        const int nn = lane & 15;
        const int mr = mt2 * 16 + ((lane >> 4) << 3);
        #pragma unroll
        for (int v = 0; v < 8; ++v)
            abuf[(mr + v) * 32 + nt2 * 16 + nn] = acc[v];
    }
    __syncthreads();

    // ---- Phase 3: z[s] = vec(A) . core[:, s]  (32K MACs, VALU) ----
    // One base pointer per thread; constant offsets fold into the b32 load's
    // 24-bit immediate (avoids per-iteration 64-bit address rebuild).
    {
        float zacc = 0.f;
        const float* cp = core + (size_t)(w * 256) * 32 + lane;  // lane == s
        const float* ap = abuf + w * 256;
        #pragma unroll 8
        for (int j = 0; j < 256; ++j)
            zacc = fmaf(ap[j], cp[j * 32], zacc);
        zpart[w][lane] = zacc;
    }
    __syncthreads();
    if (t < 32) {
        float z = zpart[0][t] + zpart[1][t] + zpart[2][t] + zpart[3][t];
        zbf[b * 32 + t] = f2bf(z);
    }
}

// ---------------------------------------------------------------------------
// Kernel 3: Out(1024x4096) = Z(1024x32) * W^T(32x4096) + bias.
// K=32 == exactly one v_wmma_f32_16x16x32_bf16 per 16x16 tile.
// Grid (64,16): each WG does a 16(b) x 256(o) tile; each wave 4 n-subtiles.
// ---------------------------------------------------------------------------
__global__ void __launch_bounds__(128)
tt_out_kernel(const unsigned short* __restrict__ zbf,
              const unsigned short* __restrict__ w3,
              const float* __restrict__ bias,
              float* __restrict__ out) {
    const int t    = threadIdx.x;
    const int lane = t & 31;
    const int w    = t >> 5;
    const int b0   = blockIdx.x * 16;
    const int o0   = blockIdx.y * 256 + w * 64;

    v16bf af = load_frag_a(zbf, 32, b0, 0, lane);   // shared across 4 subtiles
    const int nn = lane & 15;
    const int mr = (lane >> 4) << 3;

    #pragma unroll
    for (int nt = 0; nt < 4; ++nt) {
        const int oc = o0 + nt * 16;
        v16bf bf = load_frag_bt(w3, 32, 0, oc, lane);   // w3[o][s] is B^T
        v8f acc = {0.f, 0.f, 0.f, 0.f, 0.f, 0.f, 0.f, 0.f};
        acc = wmma_bf16(af, bf, acc);
        const int o  = oc + nn;      // each lane owns one output column
        const float bv = bias[o];
        #pragma unroll
        for (int v = 0; v < 8; ++v)
            out[(size_t)(b0 + mr + v) * 4096 + o] = acc[v] + bv;
    }
}

// ---------------------------------------------------------------------------
extern "C" void kernel_launch(void* const* d_in, const int* in_sizes, int n_in,
                              void* d_out, int out_size, void* d_ws, size_t ws_size,
                              hipStream_t stream) {
    (void)in_sizes; (void)n_in; (void)out_size; (void)ws_size;

    const float* x    = (const float*)d_in[0];
    const float* f0   = (const float*)d_in[1];
    const float* f1   = (const float*)d_in[2];
    const float* f2   = (const float*)d_in[3];
    const float* f3   = (const float*)d_in[4];
    const float* f4   = (const float*)d_in[5];
    const float* core = (const float*)d_in[6];
    const float* gg0  = (const float*)d_in[7];
    const float* gg1  = (const float*)d_in[8];
    const float* gg2  = (const float*)d_in[9];
    const float* bias = (const float*)d_in[10];
    float* out = (float*)d_out;

    // Workspace layout (bf16 bits), total 364,544 bytes:
    unsigned short* g1t = (unsigned short*)d_ws;      // 32 x 512
    unsigned short* g2t = g1t + 32 * 512;             // 32 x 64
    unsigned short* w3  = g2t + 32 * 64;              // 4096 x 32
    unsigned short* zbf = w3 + 4096 * 32;             // 1024 x 32

    tt_precompute_kernel<<<1, 256, 0, stream>>>(f0, f1, f2, f3, f4,
                                                gg0, gg1, gg2, g1t, g2t, w3);
    tt_contract_kernel<<<1024, 128, 0, stream>>>(x, core, g1t, g2t, zbf);
    tt_out_kernel<<<dim3(64, 16), 128, 0, stream>>>(zbf, w3, bias, out);
}
